// MultiheadAttention_7825430413514
// MI455X (gfx1250) — matrix operand order repaired and checked by reading, not compiled, not run on hardware
//
#include <hip/hip_runtime.h>
#include <hip/hip_bf16.h>

// Problem dims (compile-time constants from the reference)
#define TT 1024
#define BB 8
#define EE 1024
#define HH 16
#define HD 64
#define RR 16
#define THREE_E (3*EE)
#define MROWS (TT*BB)          // 8192

typedef __attribute__((ext_vector_type(16))) _Float16 v16h;
typedef __attribute__((ext_vector_type(8)))  float    v8f;

// ---------------------------------------------------------------------------
// WMMA fragment loaders (CDNA5 ISA §7.12.2, f16 16x16x32)
// A (16xK rows, row-major src, stride in halves):
//   lanes 0-15 : row M=lr, halves 0..7 = K k0..k0+7,   halves 8..15 = K k0+16..k0+23
//   lanes16-31 : row M=lr, halves 0..7 = K k0+8..k0+15, halves 8..15 = K k0+24..k0+31
__device__ __forceinline__ v16h load_a_frag(const _Float16* base, int stride,
                                            int m0, int k0, int lr, int hs) {
  const _Float16* p = base + (size_t)(m0 + lr) * stride + k0 + hs * 8;
  union { v16h v; uint4 q[2]; } u;
  u.q[0] = *(const uint4*)(p);        // 8 halves
  u.q[1] = *(const uint4*)(p + 16);   // 8 halves, 16 K further
  return u.v;
}
// B (Kx16): column N = lr must be contiguous over K in memory, i.e. src is
// stored [N][K] row-major (stride in halves).
//   lanes 0-15 : N=lr, halves 0..7 = K k0..k0+7,   halves 8..15 = K k0+16..k0+23
//   lanes16-31 : N=lr, halves 0..7 = K k0+8..k0+15, halves 8..15 = K k0+24..k0+31
__device__ __forceinline__ v16h load_b_frag(const _Float16* base, int stride,
                                            int n0, int k0, int lr, int hs) {
  const _Float16* p = base + (size_t)(n0 + lr) * stride + k0 + hs * 8;
  union { v16h v; uint4 q[2]; } u;
  u.q[0] = *(const uint4*)(p);
  u.q[1] = *(const uint4*)(p + 16);
  return u.v;
}

__device__ __forceinline__ v8f wmma_f16(v16h a, v16h b, v8f c) {
  return __builtin_amdgcn_wmma_f32_16x16x32_f16(false, a, false, b, (short)0, c,
                                                false, false);
}

// ---------------------------------------------------------------------------
// Ping-pong pipelined 32M x 64N per-wave GEMM core (K = EE, f16 in, f32 acc).
// Manual 2x k-unroll with two fragment sets: loads define the loop-carried
// registers directly (no buffer-rotation movs, no WMMA WAR hazard nops), and
// each WMMA group waits only on loads issued a half-iteration earlier.
template <int JT>
struct Frag {
  v16h a0, a1;
  v16h b[JT];
};

template <int JT>
__device__ __forceinline__ void load_set(Frag<JT>& s,
                                         const _Float16* __restrict__ A,
                                         const _Float16* __restrict__ Bw,
                                         int m0, int n0, int k0, int lr, int hs) {
  s.a0 = load_a_frag(A, EE, m0,      k0, lr, hs);
  s.a1 = load_a_frag(A, EE, m0 + 16, k0, lr, hs);
#pragma unroll
  for (int j = 0; j < JT; ++j) s.b[j] = load_b_frag(Bw, EE, n0 + j * 16, k0, lr, hs);
}

template <int JT>
__device__ __forceinline__ void mma_set(const Frag<JT>& s, v8f (&c)[2][JT]) {
#pragma unroll
  for (int j = 0; j < JT; ++j) {
    c[0][j] = wmma_f16(s.a0, s.b[j], c[0][j]);
    c[1][j] = wmma_f16(s.a1, s.b[j], c[1][j]);
  }
}

template <int JT>
__device__ __forceinline__ void gemm_pipe(const _Float16* __restrict__ A,
                                          const _Float16* __restrict__ Bw,
                                          int m0, int n0, int lane,
                                          v8f (&c)[2][JT]) {
  const int lr = lane & 15, hs = lane >> 4;
  Frag<JT> s0, s1;
  load_set<JT>(s0, A, Bw, m0, n0, 0, lr, hs);
  int k0 = 0;
#pragma unroll 1
  for (; k0 + 64 < EE; k0 += 64) {
    // prefetch one 128B line per row, 3 iterations ahead (uniform branch)
    if (k0 + 192 < EE) {
      const int kp = k0 + 192;
      __builtin_prefetch(A  + (size_t)(m0 + lane) * EE + kp, 0, 1);
      __builtin_prefetch(Bw + (size_t)(n0 + lane) * EE + kp, 0, 1);
      __builtin_prefetch(Bw + (size_t)(n0 + 32 + lane) * EE + kp, 0, 1);
    }
    load_set<JT>(s1, A, Bw, m0, n0, k0 + 32, lr, hs);
    mma_set<JT>(s0, c);
    load_set<JT>(s0, A, Bw, m0, n0, k0 + 64, lr, hs);
    mma_set<JT>(s1, c);
  }
  load_set<JT>(s1, A, Bw, m0, n0, k0 + 32, lr, hs);
  mma_set<JT>(s0, c);
  mma_set<JT>(s1, c);
}

// ---------------------------------------------------------------------------
// Kernel 1: effective weights (W + L*R) -> f16, and query -> f16
__global__ __launch_bounds__(256)
void mha_prep(const float* __restrict__ query,
              const float* __restrict__ inW, const float* __restrict__ inL,
              const float* __restrict__ inR,
              const float* __restrict__ outW, const float* __restrict__ outL,
              const float* __restrict__ outR,
              _Float16* __restrict__ Win16, _Float16* __restrict__ Wout16,
              _Float16* __restrict__ q16) {
  const size_t N1 = (size_t)THREE_E * EE;   // 3,145,728
  const size_t N2 = (size_t)EE * EE;        // 1,048,576
  const size_t N3 = (size_t)MROWS * EE;     // 8,388,608
  size_t idx = (size_t)blockIdx.x * 256 + threadIdx.x;
  if (idx < N1) {
    int f = (int)(idx >> 10), e = (int)(idx & 1023);
    float acc = inW[idx];
#pragma unroll
    for (int r = 0; r < RR; ++r) acc += inL[f * RR + r] * inR[r * EE + e];
    Win16[idx] = (_Float16)acc;
  } else if (idx < N1 + N2) {
    size_t j = idx - N1;
    int f = (int)(j >> 10), e = (int)(j & 1023);
    float acc = outW[j];
#pragma unroll
    for (int r = 0; r < RR; ++r) acc += outL[f * RR + r] * outR[r * EE + e];
    Wout16[j] = (_Float16)acc;
  } else if (idx < N1 + N2 + N3) {
    size_t j = idx - N1 - N2;
    q16[j] = (_Float16)query[j];
  }
}

// ---------------------------------------------------------------------------
// Kernel 2: in-projection GEMM  acti[8192][3072] = q16 * Win16^T, scatter into
// qh[b][h][t][d] (scaled), kh[b][h][t][d], vT[b][h][d][t]  (all f16)
__global__ __launch_bounds__(128)
void mha_gemm_inproj(const _Float16* __restrict__ A,     // [8192][1024]
                     const _Float16* __restrict__ Bw,    // [3072][1024]
                     const float* __restrict__ bias,     // [3072]
                     _Float16* __restrict__ qh, _Float16* __restrict__ kh,
                     _Float16* __restrict__ vT) {
  const int lane = threadIdx.x & 31, wv = threadIdx.x >> 5;
  const int lr = lane & 15, hs = lane >> 4;
  const int m0 = blockIdx.x * 32;
  const int n0 = (blockIdx.y * 4 + wv) * 64;
  v8f c[2][4];
#pragma unroll
  for (int mi = 0; mi < 2; ++mi)
#pragma unroll
    for (int j = 0; j < 4; ++j) c[mi][j] = (v8f){0,0,0,0,0,0,0,0};

  gemm_pipe<4>(A, Bw, m0, n0, lane, c);

#pragma unroll
  for (int mi = 0; mi < 2; ++mi) {
#pragma unroll
    for (int j = 0; j < 4; ++j) {
#pragma unroll
      for (int r = 0; r < 8; ++r) {
        int row = m0 + mi * 16 + hs * 8 + r;       // (t*B + b)
        int f   = n0 + j * 16 + lr;
        float val = c[mi][j][r] + bias[f];
        int t = row >> 3, b = row & 7;
        int sec = f >> 10, fe = f & 1023;
        int h = fe >> 6, d = fe & 63;
        size_t bh = (size_t)b * HH + h;
        if (sec == 0) {
          qh[(bh * TT + t) * HD + d] = (_Float16)(val * 0.125f);  // hd^-0.5
        } else if (sec == 1) {
          kh[(bh * TT + t) * HD + d] = (_Float16)val;
        } else {
          vT[(bh * HD + d) * TT + t] = (_Float16)val;             // transposed
        }
      }
    }
  }
}

// ---------------------------------------------------------------------------
// Kernel 3: flash attention.  One wave = one (b,h) x 16-query-row block.
// Computes S^T = K*Q^T so score tiles land with query index = lane; the
// exp'd tile pair is directly the 16x32 f16 A-operand for P*V.
__global__ __launch_bounds__(128)
void mha_attn(const _Float16* __restrict__ qh, const _Float16* __restrict__ kh,
              const _Float16* __restrict__ vT, const float* __restrict__ relb,
              _Float16* __restrict__ attn_out) {
  const int lane = threadIdx.x & 31, wv = threadIdx.x >> 5;
  const int lr = lane & 15, hs = lane >> 4;
  const int bh = blockIdx.y;                    // b*H + h
  const int h = bh & (HH - 1), b = bh >> 4;
  const int t0 = (blockIdx.x * 4 + wv) * 16;

  const _Float16* qbase = qh + (size_t)bh * TT * HD;
  const _Float16* kbase = kh + (size_t)bh * TT * HD;
  const _Float16* vbase = vT + (size_t)bh * HD * TT;
  const float* brow = relb + ((size_t)h * TT + (t0 + lr)) * TT;

  // Q^T B-operand fragments (fixed for this t-block): K = head dim, N = t
  v16h qb0 = load_b_frag(qbase, HD, t0, 0,  lr, hs);
  v16h qb1 = load_b_frag(qbase, HD, t0, 32, lr, hs);

  v8f z = {0,0,0,0,0,0,0,0};
  v8f o0 = z, o1 = z, o2 = z, o3 = z;
  float mrun = -1e30f, lrun = 0.0f;

  for (int s0 = 0; s0 < TT; s0 += 32) {
    // S^T tiles: A = K rows, B = Q^T
    v16h ka00 = load_a_frag(kbase, HD, s0,      0,  lr, hs);
    v16h ka01 = load_a_frag(kbase, HD, s0,      32, lr, hs);
    v16h ka10 = load_a_frag(kbase, HD, s0 + 16, 0,  lr, hs);
    v16h ka11 = load_a_frag(kbase, HD, s0 + 16, 32, lr, hs);
    v8f st0 = z, st1 = z;
    st0 = wmma_f16(ka00, qb0, st0);
    st0 = wmma_f16(ka01, qb1, st0);
    st1 = wmma_f16(ka10, qb0, st1);
    st1 = wmma_f16(ka11, qb1, st1);

    // + rel_pos_bias[h][t][s] : C-layout element (s = s0+tile*16+hs*8+r, t = t0+lr)
    const float* bp0 = brow + s0 + hs * 8;
    const float* bp1 = brow + s0 + 16 + hs * 8;
    float4 f0a = *(const float4*)(bp0), f0b = *(const float4*)(bp0 + 4);
    float4 f1a = *(const float4*)(bp1), f1b = *(const float4*)(bp1 + 4);
    st0[0] += f0a.x; st0[1] += f0a.y; st0[2] += f0a.z; st0[3] += f0a.w;
    st0[4] += f0b.x; st0[5] += f0b.y; st0[6] += f0b.z; st0[7] += f0b.w;
    st1[0] += f1a.x; st1[1] += f1a.y; st1[2] += f1a.z; st1[3] += f1a.w;
    st1[4] += f1b.x; st1[5] += f1b.y; st1[6] += f1b.z; st1[7] += f1b.w;

    // column (per-query) max over 16 regs + partner half-wave
    float mx = st0[0];
#pragma unroll
    for (int r = 1; r < 8; ++r) mx = fmaxf(mx, st0[r]);
#pragma unroll
    for (int r = 0; r < 8; ++r) mx = fmaxf(mx, st1[r]);
    mx = fmaxf(mx, __shfl_xor(mx, 16, 32));
    float mnew = fmaxf(mrun, mx);
    float alpha = __expf(mrun - mnew);

    float p0[8], p1[8], ssum = 0.0f;
#pragma unroll
    for (int r = 0; r < 8; ++r) {
      p0[r] = __expf(st0[r] - mnew); ssum += p0[r];
      p1[r] = __expf(st1[r] - mnew); ssum += p1[r];
    }
    ssum += __shfl_xor(ssum, 16, 32);
    lrun = lrun * alpha + ssum;
    mrun = mnew;

    // rescale O: row m = hs*8 + r needs alpha of query t0+hs*8+r (lives in lane hs*8+r)
#pragma unroll
    for (int r = 0; r < 8; ++r) {
      float sc = __shfl(alpha, r + hs * 8, 32);
      o0[r] *= sc; o1[r] *= sc; o2[r] *= sc; o3[r] *= sc;
    }

    // exp'd S^T tile pair == A-operand (16 t-rows x 32 s) for P*V
    v16h pa;
#pragma unroll
    for (int r = 0; r < 8; ++r) {
      pa[r]     = (_Float16)p0[r];
      pa[8 + r] = (_Float16)p1[r];
    }
    v16h vb0 = load_b_frag(vbase, TT, 0,  s0, lr, hs);
    v16h vb1 = load_b_frag(vbase, TT, 16, s0, lr, hs);
    v16h vb2 = load_b_frag(vbase, TT, 32, s0, lr, hs);
    v16h vb3 = load_b_frag(vbase, TT, 48, s0, lr, hs);
    o0 = wmma_f16(pa, vb0, o0);
    o1 = wmma_f16(pa, vb1, o1);
    o2 = wmma_f16(pa, vb2, o2);
    o3 = wmma_f16(pa, vb3, o3);
  }

  // finalize: divide row t by its softmax sum, store f16 rows for out-proj GEMM
#pragma unroll
  for (int r = 0; r < 8; ++r) {
    float li = __shfl(lrun, r + hs * 8, 32);
    float inv = 1.0f / li;
    int t = t0 + hs * 8 + r;
    size_t rowbase = ((size_t)t * BB + b) * EE + (size_t)h * HD;
    attn_out[rowbase + 0  + lr] = (_Float16)(o0[r] * inv);
    attn_out[rowbase + 16 + lr] = (_Float16)(o1[r] * inv);
    attn_out[rowbase + 32 + lr] = (_Float16)(o2[r] * inv);
    attn_out[rowbase + 48 + lr] = (_Float16)(o3[r] * inv);
  }
}

// ---------------------------------------------------------------------------
// Kernel 4: out-projection GEMM  out[8192][1024] = attn_out * Wout16^T + bias
__global__ __launch_bounds__(128)
void mha_gemm_outproj(const _Float16* __restrict__ A,   // [8192][1024]
                      const _Float16* __restrict__ Bw,  // [1024][1024]
                      const float* __restrict__ bias, float* __restrict__ out) {
  const int lane = threadIdx.x & 31, wv = threadIdx.x >> 5;
  const int lr = lane & 15, hs = lane >> 4;
  const int m0 = blockIdx.x * 32;
  const int n0 = (blockIdx.y * 4 + wv) * 64;
  v8f c[2][4];
#pragma unroll
  for (int mi = 0; mi < 2; ++mi)
#pragma unroll
    for (int j = 0; j < 4; ++j) c[mi][j] = (v8f){0,0,0,0,0,0,0,0};

  gemm_pipe<4>(A, Bw, m0, n0, lane, c);

#pragma unroll
  for (int mi = 0; mi < 2; ++mi) {
#pragma unroll
    for (int j = 0; j < 4; ++j) {
#pragma unroll
      for (int r = 0; r < 8; ++r) {
        int row = m0 + mi * 16 + hs * 8 + r;
        int f   = n0 + j * 16 + lr;
        out[(size_t)row * EE + f] = c[mi][j][r] + bias[f];
      }
    }
  }
}

// ---------------------------------------------------------------------------
extern "C" void kernel_launch(void* const* d_in, const int* in_sizes, int n_in,
                              void* d_out, int out_size, void* d_ws, size_t ws_size,
                              hipStream_t stream) {
  const float* query   = (const float*)d_in[0];
  const float* inW     = (const float*)d_in[1];
  const float* inBias  = (const float*)d_in[2];
  const float* inL     = (const float*)d_in[3];
  const float* inR     = (const float*)d_in[4];
  const float* outW    = (const float*)d_in[5];
  const float* outBias = (const float*)d_in[6];
  const float* outL    = (const float*)d_in[7];
  const float* outR    = (const float*)d_in[8];
  const float* relb    = (const float*)d_in[9];
  float* out = (float*)d_out;

  // Workspace layout (bytes)
  char* ws = (char*)d_ws;
  _Float16* Win16  = (_Float16*)(ws);                         //  6 MB  [3072][1024]
  _Float16* Wout16 = (_Float16*)(ws + 6291456);               //  2 MB  [1024][1024]
  _Float16* qh     = (_Float16*)(ws + 8388608);               // 16 MB  [b][h][t][d]
  _Float16* kh     = (_Float16*)(ws + 25165824);              // 16 MB  [b][h][t][d]
  _Float16* vT     = (_Float16*)(ws + 41943040);              // 16 MB  [b][h][d][t]
  _Float16* q16    = (_Float16*)(ws + 58720256);              // 16 MB  [8192][1024]
  _Float16* attn_o = q16;  // reuse: q16 dead after in-proj GEMM

  // 1) effective weights + f16 query  (12,582,912 threads)
  mha_prep<<<49152, 256, 0, stream>>>(query, inW, inL, inR, outW, outL, outR,
                                      Win16, Wout16, q16);
  // 2) in-projection GEMM + head scatter: M=8192, N=3072 (wave = 32x64 tile)
  mha_gemm_inproj<<<dim3(MROWS / 32, THREE_E / 256), 128, 0, stream>>>(
      q16, Win16, inBias, qh, kh, vT);
  // 3) attention: 64 t-blocks x 128 (b,h); 4 waves/block
  mha_attn<<<dim3(16, BB * HH), 128, 0, stream>>>(qh, kh, vT, relb, attn_o);
  // 4) out-projection GEMM: M=8192, N=1024 (wave = 32x64 tile)
  mha_gemm_outproj<<<dim3(MROWS / 32, EE / 256), 128, 0, stream>>>(
      attn_o, Wout16, outBias, out);
}
